// NewsTransformer_68719476736144
// MI455X (gfx1250) — compile-verified, matrix operand order checked by reading, not run on hardware
//
#include <hip/hip_runtime.h>

#define B_  512
#define S_  32
#define E_  768
#define H_  12
#define HD_ 64
#define TD_ 128
#define AH_ 128

typedef __attribute__((ext_vector_type(16))) __bf16 v16bf;
typedef __attribute__((ext_vector_type(2)))  __bf16 v2bf;
typedef __attribute__((ext_vector_type(8)))  float  v8f;

__device__ __forceinline__ __bf16 f2bf(float f) { return (__bf16)f; }

// 16-bit WMMA fragment layout helpers (ISA 7.12.2, 16x32 A / 32x16 B):
// lane -> matrix row (A) / col (B); element pair p -> base K index.
__device__ __forceinline__ int frag_row(int lane) { return lane & 15; }
__device__ __forceinline__ int frag_kpair(int lane, int p) {
  int hi = (lane >> 4) & 1;
  int base = (p < 4) ? (2 * p) : (16 + 2 * (p - 4));
  return base + (hi ? 8 : 0);
}

__device__ __forceinline__ v8f wmma_bf16(v16bf a, v16bf b, v8f c) {
  return __builtin_amdgcn_wmma_f32_16x16x32_bf16(false, a, false, b, (short)0, c,
                                                 false, false);
}

// ---------------- Kernel 1: embedding gather + cast to bf16 -----------------
__global__ void k_gather_x(const int* __restrict__ tok,
                           const float* __restrict__ wemb,
                           __bf16* __restrict__ xbf) {
  const long total = (long)B_ * S_ * E_;
  for (long i = (long)blockIdx.x * blockDim.x + threadIdx.x; i < total;
       i += (long)gridDim.x * blockDim.x) {
    long row = i / E_;                 // b*S + s
    int  e   = (int)(i - row * E_);
    int  t   = tok[row];
    float v  = (t == 0) ? 0.0f : wemb[(long)t * E_ + e];   // padding_idx = 0
    xbf[i]   = f2bf(v);
  }
}

// ------- Kernel 2: pack an f32 row-major [rows, ld] matrix into WMMA --------
// ------- B-fragment order: frag fi = rowTile*kTiles + kTile, 512 bf16 -------
__global__ void k_pack_b(const float* __restrict__ src, __bf16* __restrict__ dst,
                         int rowTiles, int kTiles, int ld) {
  const int lane = threadIdx.x & 31;
  const int wave = threadIdx.x >> 5;
  const int fi   = blockIdx.x * (blockDim.x >> 5) + wave;
  if (fi >= rowTiles * kTiles) return;
  const int kt = fi % kTiles;
  const int nt = fi / kTiles;
  const float* s = src + (long)(nt * 16 + frag_row(lane)) * ld + kt * 32;
  __bf16*      d = dst + (long)fi * 512 + lane * 16;
#pragma unroll
  for (int p = 0; p < 8; ++p) {
    int k0    = frag_kpair(lane, p);
    d[2*p]    = f2bf(s[k0]);
    d[2*p+1]  = f2bf(s[k0 + 1]);
  }
}

// --------- Kernel 3: fused per-(b) attention: projection -> scores ----------
// --------- -> softmax -> head_out, looping over the 12 heads ----------------
__global__ void __launch_bounds__(512)
k_attn(const __bf16* __restrict__ xbf, const __bf16* __restrict__ wpack,
       float* __restrict__ attended) {
  __shared__ __bf16 Tlds[S_][E_ + 8];     // 48.5 KB, padded vs bank conflicts
  __shared__ float  slds[S_][S_ + 1];     // 4.1 KB
  __shared__ __bf16 alds[S_][S_ + 2];     // 2.1 KB

  const int b    = blockIdx.x;
  const int tid  = threadIdx.x;
  const int lane = tid & 31;
  const int wave = tid >> 5;
  const int hi   = (lane >> 4) & 1;
  const int fr   = lane & 15;

  const __bf16* xb = xbf + (size_t)b * S_ * E_;

  for (int h = 0; h < H_; ++h) {
    // ---- GEMM1: T = x_b @ W_h^T   (M=32, N=768, K=768) ----
    // wave w owns mt = w&1, nt = (w>>1) + 8*i (i = 0..5): 6 output tiles.
    const int mt  = wave & 1;
    const int nt0 = wave >> 1;
    v8f zero = {};
    v8f acc[6];
#pragma unroll
    for (int i = 0; i < 6; ++i) acc[i] = zero;
    for (int kt = 0; kt < 24; ++kt) {
      v16bf a;
      const __bf16* ap = xb + (size_t)(mt * 16 + fr) * E_ + kt * 32;
#pragma unroll
      for (int p = 0; p < 8; ++p) {
        int k0 = frag_kpair(lane, p);
        v2bf t = *(const v2bf*)(ap + k0);
        a[2*p] = t[0]; a[2*p+1] = t[1];
      }
#pragma unroll
      for (int i = 0; i < 6; ++i) {
        int nt = nt0 + 8 * i;
        const v16bf bfrag = *(const v16bf*)(
            wpack + ((size_t)((h * 48 + nt) * 24 + kt)) * 512 + lane * 16);
        acc[i] = wmma_bf16(a, bfrag, acc[i]);
      }
    }
#pragma unroll
    for (int i = 0; i < 6; ++i) {
      int nt = nt0 + 8 * i;
#pragma unroll
      for (int r = 0; r < 8; ++r)
        Tlds[mt * 16 + hi * 8 + r][nt * 16 + fr] = f2bf(acc[i][r]);
    }
    __syncthreads();

    // ---- scores = T @ x_b^T   (M=32, N=32, K=768), waves 0..3 ----
    if (wave < 4) {
      const int smt = wave & 1, snt = wave >> 1;
      v8f sa = zero;
      for (int kt = 0; kt < 24; ++kt) {
        v16bf a, bb;
        const __bf16* ta = &Tlds[smt * 16 + fr][kt * 32];
        const __bf16* tb = xb + (size_t)(snt * 16 + fr) * E_ + kt * 32;
#pragma unroll
        for (int p = 0; p < 8; ++p) {
          int k0 = frag_kpair(lane, p);
          v2bf u = *(const v2bf*)(ta + k0);
          a[2*p]  = u[0]; a[2*p+1]  = u[1];
          v2bf w = *(const v2bf*)(tb + k0);
          bb[2*p] = w[0]; bb[2*p+1] = w[1];
        }
        sa = wmma_bf16(a, bb, sa);
      }
#pragma unroll
      for (int r = 0; r < 8; ++r)
        slds[smt * 16 + hi * 8 + r][snt * 16 + fr] = sa[r];
    }
    __syncthreads();

    // ---- row softmax over t (one thread per s-row) ----
    if (tid < S_) {
      float mx = -3.4e38f;
      for (int t = 0; t < S_; ++t) mx = fmaxf(mx, slds[tid][t]);
      float sum = 0.f;
      for (int t = 0; t < S_; ++t) sum += __expf(slds[tid][t] - mx);
      float inv = 1.0f / sum;
      for (int t = 0; t < S_; ++t)
        alds[tid][t] = f2bf(__expf(slds[tid][t] - mx) * inv);
    }
    __syncthreads();

    // ---- head_out = attn @ x_b[:, :64]  (M=32, N=64, K=32), waves 0..7 ----
    if (wave < 8) {
      const int hmt = wave & 1, hnt = wave >> 1;   // hnt = 0..3
      v16bf a, bb;
#pragma unroll
      for (int p = 0; p < 8; ++p) {
        int k0 = frag_kpair(lane, p);
        v2bf u = *(const v2bf*)(&alds[hmt * 16 + fr][k0]);
        a[2*p] = u[0]; a[2*p+1] = u[1];
        // B[k][n] = x_b[t=k][d = hnt*16 + fr]  (column gather)
        bb[2*p]   = xb[(size_t)k0       * E_ + hnt * 16 + fr];
        bb[2*p+1] = xb[(size_t)(k0 + 1) * E_ + hnt * 16 + fr];
      }
      v8f ho = zero;
      ho = wmma_bf16(a, bb, ho);
      float* outp = attended + (size_t)b * S_ * E_;
#pragma unroll
      for (int r = 0; r < 8; ++r)
        outp[(size_t)(hmt * 16 + hi * 8 + r) * E_ + h * HD_ + hnt * 16 + fr] =
            ho[r];
    }
    __syncthreads();
  }
}

// -------- Kernel 4: additive-attention pooling + topic concat per b ---------
__global__ void __launch_bounds__(512)
k_pool(const float* __restrict__ attended, const __bf16* __restrict__ uwpack,
       const float* __restrict__ uw_b, const float* __restrict__ u_w,
       const float* __restrict__ q_w, const int* __restrict__ topic_id,
       const float* __restrict__ topic_emb, float* __restrict__ out) {
  __shared__ float ulds[S_][AH_ + 1];   // 16.5 KB
  __shared__ float alpha[S_];

  const int b    = blockIdx.x;
  const int tid  = threadIdx.x;
  const int lane = tid & 31;
  const int wave = tid >> 5;
  const int hi   = (lane >> 4) & 1;
  const int fr   = lane & 15;
  const float* att = attended + (size_t)b * S_ * E_;

  // u_pre = attended @ Uw^T : M=32, N=128, K=768; 16 tiles, one per wave.
  const int mt = wave & 1, nt = wave >> 1;   // nt = 0..7
  v8f zero = {};
  v8f acc = zero;
  for (int kt = 0; kt < 24; ++kt) {
    v16bf a;
    const float* ap = att + (size_t)(mt * 16 + fr) * E_ + kt * 32;
#pragma unroll
    for (int p = 0; p < 8; ++p) {
      int k0 = frag_kpair(lane, p);
      a[2*p]   = f2bf(ap[k0]);
      a[2*p+1] = f2bf(ap[k0 + 1]);
    }
    const v16bf bfrag =
        *(const v16bf*)(uwpack + ((size_t)(nt * 24 + kt)) * 512 + lane * 16);
    acc = wmma_bf16(a, bfrag, acc);
  }
  {
    int n = nt * 16 + fr;
    float bias = uw_b[n] + u_w[n];
#pragma unroll
    for (int r = 0; r < 8; ++r)
      ulds[mt * 16 + hi * 8 + r][n] = tanhf(acc[r] + bias);
  }
  __syncthreads();

  if (tid < S_) {                 // logits = u @ q_w
    float dot = 0.f;
    for (int a2 = 0; a2 < AH_; ++a2) dot += ulds[tid][a2] * q_w[a2];
    alpha[tid] = dot;
  }
  __syncthreads();
  if (tid == 0) {                 // softmax over s (S=32, serial is cheap)
    float mx = -3.4e38f;
    for (int s = 0; s < S_; ++s) mx = fmaxf(mx, alpha[s]);
    float sum = 0.f;
    for (int s = 0; s < S_; ++s) sum += __expf(alpha[s] - mx);
    float inv = 1.0f / sum;
    for (int s = 0; s < S_; ++s) alpha[s] = __expf(alpha[s] - mx) * inv;
  }
  __syncthreads();

  float* ob = out + (size_t)b * (E_ + TD_);
  for (int e = tid; e < E_; e += blockDim.x) {   // title_repr
    float s = 0.f;
    for (int t = 0; t < S_; ++t) s += alpha[t] * att[(size_t)t * E_ + e];
    ob[e] = s;
  }
  const int trow = topic_id[b];
  for (int d = tid; d < TD_; d += blockDim.x)    // topic_repr concat
    ob[E_ + d] = topic_emb[(size_t)trow * TD_ + d];
}

extern "C" void kernel_launch(void* const* d_in, const int* in_sizes, int n_in,
                              void* d_out, int out_size, void* d_ws,
                              size_t ws_size, hipStream_t stream) {
  const int*   title_tokens = (const int*)d_in[0];
  const int*   topic_id     = (const int*)d_in[1];
  const float* word_emb     = (const float*)d_in[2];
  const float* topic_emb    = (const float*)d_in[3];
  const float* W_heads      = (const float*)d_in[4];
  const float* Uw_W         = (const float*)d_in[5];
  const float* Uw_b         = (const float*)d_in[6];
  const float* u_w          = (const float*)d_in[7];
  const float* q_w          = (const float*)d_in[8];
  float* out = (float*)d_out;

  char* ws = (char*)d_ws;
  __bf16* xbf      = (__bf16*)ws;  ws += (size_t)B_ * S_ * E_ * 2;
  __bf16* wpack    = (__bf16*)ws;  ws += (size_t)H_ * E_ * E_ * 2;
  __bf16* uwpack   = (__bf16*)ws;  ws += (size_t)AH_ * E_ * 2;
  float*  attended = (float*)ws;   ws += (size_t)B_ * S_ * E_ * 4;

  k_gather_x<<<4096, 256, 0, stream>>>(title_tokens, word_emb, xbf);
  // W_heads viewed as [H*768, 768] -> 576 row-tiles x 24 k-tiles = 13824 frags
  k_pack_b<<<(H_ * 48 * 24 + 15) / 16, 512, 0, stream>>>(W_heads, wpack,
                                                         H_ * 48, 24, E_);
  k_pack_b<<<(8 * 24 + 15) / 16, 512, 0, stream>>>(Uw_W, uwpack, 8, 24, E_);
  k_attn<<<B_, 512, 0, stream>>>(xbf, wpack, attended);
  k_pool<<<B_, 512, 0, stream>>>(attended, uwpack, Uw_b, u_w, q_w, topic_id,
                                 topic_emb, out);
}
